// BiMambaBlock_41815801594434
// MI455X (gfx1250) — matrix operand order repaired and checked by reading, not compiled, not run on hardware
//
#include <hip/hip_runtime.h>
#include <hip/hip_bf16.h>
#include <math.h>

#define B_      2
#define L_      1024
#define D_MODEL 1024
#define D_STATE 16
#define D_CONV  4
#define D_INNER 2048
#define DT_RANK 64
#define NROWS   (B_ * L_)                 // 2048
#define DBL_W   (DT_RANK + 2 * D_STATE)   // 96

typedef __attribute__((ext_vector_type(16))) __bf16 v16bf;
typedef __attribute__((ext_vector_type(8)))  __bf16 v8bf;
typedef __attribute__((ext_vector_type(8)))  float  v8f;
typedef __attribute__((ext_vector_type(4)))  int    v4i;

// pointer flavors for the async global->LDS builtin (AS1 = global, AS3 = LDS)
typedef __attribute__((address_space(1))) v4i* as1_v4i_ptr;
typedef __attribute__((address_space(3))) v4i* as3_v4i_ptr;

__device__ __forceinline__ __bf16 f2bf(float f) { return (__bf16)f; }

// -------- optional gfx1250 async global->LDS path (ASYNCcnt tracked) --------
#if defined(__has_builtin)
#  if __has_builtin(__builtin_amdgcn_global_load_async_to_lds_b128) && \
      __has_builtin(__builtin_amdgcn_s_wait_asynccnt)
#    define HAVE_ASYNC_LDS 1
#  endif
#endif
#ifndef HAVE_ASYNC_LDS
#  define HAVE_ASYNC_LDS 0
#endif

// ---------------------------------------------------------------------------
// GEMM: C[M,N] = A[M,K] * W[N,K]^T, fp32 in/out, bf16 WMMA compute.
// Requirements satisfied by every call site: M % 64 == 0, K % 32 == 0,
// grid.y == M/64 (so no A-row guard), only N may be ragged vs the 128 tile.
// Block: 256 threads = 8 waves, 2(M) x 4(N); wave tile 32x32 = 2x2 WMMA.
// ---------------------------------------------------------------------------
__global__ __launch_bounds__(256)
void gemm_bf16_wmma(const float* __restrict__ A, int lda, int acol0,
                    const float* __restrict__ W, int ldw,
                    float* __restrict__ C0, float* __restrict__ C1,
                    int ldc, int splitN,
                    int M, int N, int K,
                    int flip, int Lseq,
                    const float* __restrict__ bias, int epilogue)
{
    __shared__ __bf16 As[64][40];    // 64x32 tile (stride 40 -> 80B rows, 16B aligned)
    __shared__ __bf16 Bs[128][40];   // 128x32 tile

    const int tid   = threadIdx.x;
    const int lane  = tid & 31;
    const int wave  = tid >> 5;
    const int waveM = wave & 1;      // 2 waves in M, 32 rows each
    const int waveN = wave >> 1;     // 4 waves in N, 32 cols each
    const int blockM = blockIdx.y * 64;
    const int blockN = blockIdx.x * 128;

    // ---- staging geometry, hoisted out of the K loop ----
    // A tile: thread covers row (tid>>2), cols [(tid&3)*8, +8)
    const int arow = tid >> 2;
    const int acol = (tid & 3) << 3;
    int prowA = blockM + arow;
    if (flip) { int bb = prowA / Lseq, ll = prowA % Lseq; prowA = bb * Lseq + (Lseq - 1 - ll); }
    const float* __restrict__ Aptr = A + (size_t)prowA * lda + acol0 + acol;

    // B tile: thread covers row (tid>>1), cols [(tid&1)*16, +16)
    const int brow = tid >> 1;
    const int bcol = (tid & 1) << 4;
    const int gnB  = blockN + brow;
    const bool bvalid = (gnB < N);
    const float* __restrict__ Wptr = W + (size_t)gnB * ldw + bcol;

    v8f acc[2][2];
#pragma unroll
    for (int mi = 0; mi < 2; ++mi)
#pragma unroll
        for (int nj = 0; nj < 2; ++nj)
#pragma unroll
            for (int i = 0; i < 8; ++i) acc[mi][nj][i] = 0.0f;

    for (int k0 = 0; k0 < K; k0 += 32) {
        // ---- stage A: 8 floats via two b128 loads, packed bf16 store ----
        {
            float4 a0 = *(const float4*)(Aptr + k0);
            float4 a1 = *(const float4*)(Aptr + k0 + 4);
            v8bf p;
            p[0] = f2bf(a0.x); p[1] = f2bf(a0.y); p[2] = f2bf(a0.z); p[3] = f2bf(a0.w);
            p[4] = f2bf(a1.x); p[5] = f2bf(a1.y); p[6] = f2bf(a1.z); p[7] = f2bf(a1.w);
            *(v8bf*)&As[arow][acol] = p;
        }
        // ---- stage B: 16 floats via four b128 loads ----
        {
            float4 t0, t1, t2, t3;
            if (bvalid) {
                t0 = *(const float4*)(Wptr + k0);
                t1 = *(const float4*)(Wptr + k0 + 4);
                t2 = *(const float4*)(Wptr + k0 + 8);
                t3 = *(const float4*)(Wptr + k0 + 12);
            } else {
                t0 = t1 = t2 = t3 = make_float4(0.f, 0.f, 0.f, 0.f);
            }
            v16bf p;
            p[0]  = f2bf(t0.x); p[1]  = f2bf(t0.y); p[2]  = f2bf(t0.z); p[3]  = f2bf(t0.w);
            p[4]  = f2bf(t1.x); p[5]  = f2bf(t1.y); p[6]  = f2bf(t1.z); p[7]  = f2bf(t1.w);
            p[8]  = f2bf(t2.x); p[9]  = f2bf(t2.y); p[10] = f2bf(t2.z); p[11] = f2bf(t2.w);
            p[12] = f2bf(t3.x); p[13] = f2bf(t3.y); p[14] = f2bf(t3.z); p[15] = f2bf(t3.w);
            *(v16bf*)&Bs[brow][bcol] = p;
        }
        // speculative prefetch of next K slice (global_prefetch_b8)
        if (k0 + 32 < K) {
            __builtin_prefetch(Aptr + k0 + 32, 0, 1);
            if (bvalid) __builtin_prefetch(Wptr + k0 + 32, 0, 1);
        }
        __syncthreads();

        // ---- fragments (contiguous k = 16*(lane/16) + s, same in A and B) + 2x2 WMMA ----
        v16bf af[2], bf[2];
#pragma unroll
        for (int mi = 0; mi < 2; ++mi) {
            int r  = (waveM << 5) + (mi << 4) + (lane & 15);
            int kb = (lane >> 4) << 4;
#pragma unroll
            for (int i = 0; i < 8; ++i) af[mi][i]     = As[r][kb + i];
#pragma unroll
            for (int i = 0; i < 8; ++i) af[mi][8 + i] = As[r][kb + 8 + i];
        }
#pragma unroll
        for (int nj = 0; nj < 2; ++nj) {
            int c  = (waveN << 5) + (nj << 4) + (lane & 15);
            int kb = (lane >> 4) << 4;
#pragma unroll
            for (int i = 0; i < 16; ++i) bf[nj][i] = Bs[c][kb + i];
        }
#pragma unroll
        for (int mi = 0; mi < 2; ++mi)
#pragma unroll
            for (int nj = 0; nj < 2; ++nj)
                acc[mi][nj] = __builtin_amdgcn_wmma_f32_16x16x32_bf16(
                    false, af[mi], false, bf[nj], (short)0, acc[mi][nj], false, false);
        __syncthreads();
    }

    // ---- store: elem i, half h -> row = h*8+i, col = lane&15 ----
    const int half = lane >> 4;
    const int cn   = lane & 15;
#pragma unroll
    for (int mi = 0; mi < 2; ++mi) {
#pragma unroll
        for (int nj = 0; nj < 2; ++nj) {
            int gc = blockN + (waveN << 5) + (nj << 4) + cn;
            if (gc >= N) continue;
            float bval = (epilogue == 1) ? bias[gc] : 0.0f;
#pragma unroll
            for (int i = 0; i < 8; ++i) {
                int gr = blockM + (waveM << 5) + (mi << 4) + half * 8 + i;
                float v = acc[mi][nj][i];
                if (epilogue == 1) {
                    v += bval;
                    v = (v > 20.0f) ? v : log1pf(__expf(v));
                }
                if (gc < splitN) C0[(size_t)gr * ldc + gc] = v;
                else             C1[(size_t)gr * ldc + (gc - splitN)] = v;
            }
        }
    }
}

// ---------------------------------------------------------------------------
// Depthwise causal conv (width 4) + bias + SiLU.
// ---------------------------------------------------------------------------
__global__ __launch_bounds__(256)
void conv_silu_kernel(const float* __restrict__ xi,
                      const float* __restrict__ conv_w,
                      const float* __restrict__ conv_b,
                      float* __restrict__ xc)
{
    int idx = blockIdx.x * 256 + threadIdx.x;      // over NROWS * D_INNER
    int d  = idx & (D_INNER - 1);
    int bl = idx >> 11;
    int l  = bl & (L_ - 1);
    float acc = conv_b[d];
#pragma unroll
    for (int k = 0; k < D_CONV; ++k) {
        int ls = l - (D_CONV - 1) + k;
        if (ls >= 0)
            acc += xi[(size_t)(bl - l + ls) * D_INNER + d] * conv_w[d * D_CONV + k];
    }
    float sig = 1.0f / (1.0f + __expf(-acc));
    xc[(size_t)bl * D_INNER + d] = acc * sig;
}

// ---------------------------------------------------------------------------
// Selective scan: sequential over L, parallel over (b, d).
// B/C vectors (32 floats, 16B aligned at dbl[row*96+64]) are staged into a
// double-buffered LDS slot — via async global->LDS DMA when available.
// ---------------------------------------------------------------------------
__global__ __launch_bounds__(256)
void scan_kernel(const float* __restrict__ dbl,
                 const float* __restrict__ dt,
                 const float* __restrict__ xc,
                 const float* __restrict__ z,
                 const float* __restrict__ A_log,
                 const float* __restrict__ Dp,
                 float* __restrict__ ys)
{
    const int d   = blockIdx.x * 256 + threadIdx.x;
    const int b   = blockIdx.y;
    const int tid = threadIdx.x;

    float Acoef[D_STATE];
#pragma unroll
    for (int n = 0; n < D_STATE; ++n)
        Acoef[n] = -__expf(A_log[d * D_STATE + n]);
    const float dval = Dp[d];

    float h[D_STATE];
#pragma unroll
    for (int n = 0; n < D_STATE; ++n) h[n] = 0.0f;

    __shared__ float BC[2][32];   // [buf][0..15]=B, [16..31]=C

#if HAVE_ASYNC_LDS
    // prologue: async-stage l=0 into buffer 0 (8 lanes x b128 = 32 floats)
    if (tid < 8) {
        const float* g = dbl + (size_t)(b * L_) * DBL_W + DT_RANK + tid * 4;
        __builtin_amdgcn_global_load_async_to_lds_b128(
            (as1_v4i_ptr)g, (as3_v4i_ptr)&BC[0][tid * 4], 0, 0);
    }
    for (int l = 0; l < L_; ++l) {
        const int row = b * L_ + l;
        __builtin_amdgcn_s_wait_asynccnt(0);       // no-op for non-issuing waves
        __syncthreads();                           // publish BC[l&1]
        const float* Bv = &BC[l & 1][0];
        const float* Cv = &BC[l & 1][16];
        if (l + 1 < L_ && tid < 8) {               // async-stage next timestep
            const float* g = dbl + (size_t)(row + 1) * DBL_W + DT_RANK + tid * 4;
            __builtin_amdgcn_global_load_async_to_lds_b128(
                (as1_v4i_ptr)g, (as3_v4i_ptr)&BC[(l + 1) & 1][tid * 4], 0, 0);
        }
        float dtv = dt[(size_t)row * D_INNER + d];
        float xcv = xc[(size_t)row * D_INNER + d];
        float zv  = z[(size_t)row * D_INNER + d];
        float dtxc = dtv * xcv;
        float y = 0.0f;
#pragma unroll
        for (int n = 0; n < D_STATE; ++n) {
            float dA = __expf(dtv * Acoef[n]);
            h[n] = dA * h[n] + dtxc * Bv[n];
            y += h[n] * Cv[n];
        }
        y += xcv * dval;
        y *= zv / (1.0f + __expf(-zv));
        ys[(size_t)row * D_INNER + d] = y;
    }
#else
    for (int l = 0; l < L_; ++l) {
        const int row = b * L_ + l;
        if (tid < 32) BC[0][tid] = dbl[(size_t)row * DBL_W + DT_RANK + tid];
        __syncthreads();
        float dtv = dt[(size_t)row * D_INNER + d];
        float xcv = xc[(size_t)row * D_INNER + d];
        float zv  = z[(size_t)row * D_INNER + d];
        float dtxc = dtv * xcv;
        float y = 0.0f;
#pragma unroll
        for (int n = 0; n < D_STATE; ++n) {
            float dA = __expf(dtv * Acoef[n]);
            h[n] = dA * h[n] + dtxc * BC[0][n];
            y += h[n] * BC[0][16 + n];
        }
        y += xcv * dval;
        y *= zv / (1.0f + __expf(-zv));
        ys[(size_t)row * D_INNER + d] = y;
        __syncthreads();
    }
#endif
}

// ---------------------------------------------------------------------------
// yo = yf[b,l,:] + yb[b,L-1-l,:]; LayerNorm over D_MODEL; write output.
// ---------------------------------------------------------------------------
__global__ __launch_bounds__(256)
void combine_ln_kernel(const float* __restrict__ yf,
                       const float* __restrict__ yb,
                       const float* __restrict__ gamma,
                       const float* __restrict__ beta,
                       float* __restrict__ out)
{
    const int bl = blockIdx.x;
    const int b = bl / L_, l = bl % L_;
    const int rf = bl;
    const int rb = b * L_ + (L_ - 1 - l);

    float v[4];
    float s = 0.0f, ss = 0.0f;
#pragma unroll
    for (int i = 0; i < 4; ++i) {
        int c = threadIdx.x + i * 256;
        float t = yf[(size_t)rf * D_MODEL + c] + yb[(size_t)rb * D_MODEL + c];
        v[i] = t; s += t; ss += t * t;
    }
#pragma unroll
    for (int off = 16; off > 0; off >>= 1) {
        s  += __shfl_down(s, off, 32);
        ss += __shfl_down(ss, off, 32);
    }
    __shared__ float rs[8], rss[8];
    int lane = threadIdx.x & 31, wv = threadIdx.x >> 5;
    if (lane == 0) { rs[wv] = s; rss[wv] = ss; }
    __syncthreads();
    if (threadIdx.x == 0) {
        float ts = 0.0f, tss = 0.0f;
        for (int i = 0; i < 8; ++i) { ts += rs[i]; tss += rss[i]; }
        rs[0] = ts; rss[0] = tss;
    }
    __syncthreads();
    float mean = rs[0] * (1.0f / D_MODEL);
    float var  = rss[0] * (1.0f / D_MODEL) - mean * mean;
    float rstd = rsqrtf(var + 1e-5f);
#pragma unroll
    for (int i = 0; i < 4; ++i) {
        int c = threadIdx.x + i * 256;
        out[(size_t)rf * D_MODEL + c] = gamma[c] * (v[i] - mean) * rstd + beta[c];
    }
}

// ---------------------------------------------------------------------------
extern "C" void kernel_launch(void* const* d_in, const int* in_sizes, int n_in,
                              void* d_out, int out_size, void* d_ws, size_t ws_size,
                              hipStream_t stream)
{
    (void)in_sizes; (void)n_in; (void)out_size; (void)ws_size;

    const float* x     = (const float*)d_in[0];
    const float* gamma = (const float*)d_in[19];
    const float* beta  = (const float*)d_in[20];

    float* ws = (float*)d_ws;
    size_t off = 0;
    float* xi  = ws + off; off += (size_t)NROWS * D_INNER;   // reused as ys after conv
    float* z   = ws + off; off += (size_t)NROWS * D_INNER;
    float* xc  = ws + off; off += (size_t)NROWS * D_INNER;
    float* dbl = ws + off; off += (size_t)NROWS * DBL_W;
    float* dt  = ws + off; off += (size_t)NROWS * D_INNER;
    float* y0  = ws + off; off += (size_t)NROWS * D_MODEL;
    float* y1  = ws + off; off += (size_t)NROWS * D_MODEL;
    float* ydir[2] = { y0, y1 };
    float* ys = xi;   // xi dead after conv; alias for scan output

    for (int dir = 0; dir < 2; ++dir) {
        const float* Win  = (const float*)d_in[1 + 9 * dir + 0];  // (4096,1024)
        const float* cw   = (const float*)d_in[1 + 9 * dir + 1];  // (2048,4)
        const float* cb   = (const float*)d_in[1 + 9 * dir + 2];  // (2048,)
        const float* Wx   = (const float*)d_in[1 + 9 * dir + 3];  // (96,2048)
        const float* Wdt  = (const float*)d_in[1 + 9 * dir + 4];  // (2048,64)
        const float* bdt  = (const float*)d_in[1 + 9 * dir + 5];  // (2048,)
        const float* Alog = (const float*)d_in[1 + 9 * dir + 6];  // (2048,16)
        const float* Dp   = (const float*)d_in[1 + 9 * dir + 7];  // (2048,)
        const float* Wout = (const float*)d_in[1 + 9 * dir + 8];  // (1024,2048)

        // in_proj: [2048,4096] = x @ Win^T ; split -> xi | z
        gemm_bf16_wmma<<<dim3((2 * D_INNER) / 128, NROWS / 64), 256, 0, stream>>>(
            x, D_MODEL, 0, Win, D_MODEL, xi, z, D_INNER, D_INNER,
            NROWS, 2 * D_INNER, D_MODEL, dir, L_, nullptr, 0);

        conv_silu_kernel<<<(NROWS * D_INNER) / 256, 256, 0, stream>>>(xi, cw, cb, xc);

        // x_proj: [2048,96] = xc @ Wx^T
        gemm_bf16_wmma<<<dim3(1, NROWS / 64), 256, 0, stream>>>(
            xc, D_INNER, 0, Wx, D_INNER, dbl, nullptr, DBL_W, 1 << 30,
            NROWS, DBL_W, D_INNER, 0, L_, nullptr, 0);

        // dt_proj + softplus: [2048,2048] = dbl[:, :64] @ Wdt^T + bdt
        gemm_bf16_wmma<<<dim3(D_INNER / 128, NROWS / 64), 256, 0, stream>>>(
            dbl, DBL_W, 0, Wdt, DT_RANK, dt, nullptr, D_INNER, 1 << 30,
            NROWS, D_INNER, DT_RANK, 0, L_, bdt, 1);

        scan_kernel<<<dim3(D_INNER / 256, B_), 256, 0, stream>>>(
            dbl, dt, xc, z, Alog, Dp, ys);

        // out_proj: [2048,1024] = ys @ Wout^T
        gemm_bf16_wmma<<<dim3(D_MODEL / 128, NROWS / 64), 256, 0, stream>>>(
            ys, D_INNER, 0, Wout, D_INNER, ydir[dir], nullptr, D_MODEL, 1 << 30,
            NROWS, D_MODEL, D_INNER, 0, L_, nullptr, 0);
    }

    combine_ln_kernel<<<NROWS, 256, 0, stream>>>(y0, y1, gamma, beta, (float*)d_out);
}
